// ViT_64665027609108
// MI455X (gfx1250) — matrix operand order repaired and checked by reading, not compiled, not run on hardware
//
#include <hip/hip_runtime.h>
#include <hip/hip_bf16.h>
#include <math.h>

typedef __attribute__((ext_vector_type(16))) _Float16 v16h;
typedef __attribute__((ext_vector_type(8)))  _Float16 h8;
typedef __attribute__((ext_vector_type(4)))  _Float16 h4;
typedef __attribute__((ext_vector_type(8)))  float    v8f;

constexpr int kDepth = 6, kHeads = 6, kDim = 384, kHD = 64;
constexpr int kGrid = 32, kTok = 1024, kBatch = 4, kM = 4096;   // kM = tokens total
constexpr int kQKVN = 3 * kDim;                                  // 1152
constexpr int kWSz = 14, kNWin = 36, kWN = 196;                  // window size / count / tokens
constexpr float kScale = 0.125f;                                 // 64^-0.5
constexpr int kLd = 40;                                          // LDS stride (halves); 80B = 5x16B

// ---- workspace layout (float offsets). Total ~31.7M floats (~127 MB). ----
constexpr size_t OFF_POS  = 0;                       // 32*32*384
constexpr size_t OFF_IM2  = OFF_POS  + 393216;       // 4096*768
constexpr size_t OFF_Y    = OFF_IM2  + 3145728;      // 4096*384
constexpr size_t OFF_NRM  = OFF_Y    + 1572864;      // 4096*384
constexpr size_t OFF_QKV  = OFF_NRM  + 1572864;      // 4096*1152
constexpr size_t OFF_QKVW = OFF_QKV  + 4718592;      // 7056*1152
constexpr size_t OFF_BH   = OFF_QKVW + 8128512;      // 4*6*1024*32 max
constexpr size_t OFF_BW   = OFF_BH   + 786432;
constexpr size_t OFF_AO   = OFF_BW   + 786432;       // 4096*384
constexpr size_t OFF_AOW  = OFF_AO   + 1572864;      // 7056*384
constexpr size_t OFF_H    = OFF_AOW  + 2709504;      // 4096*1536

// -------------------------- WMMA helpers --------------------------
static __device__ __forceinline__ v8f wmma_f16(v16h a, v16h b, v8f c) {
  return __builtin_amdgcn_wmma_f32_16x16x32_f16(false, a, false, b, (short)0, c,
                                                false, false);
}

// A fragment 16x32 f16 from LDS rows at stride ld. Per the ISA layout the
// per-lane data is two contiguous 8-half runs: [kb, kb+8) and [kb+16, kb+24),
// kb = (lane>=16)*8  ->  two ds_load_b128.
static __device__ __forceinline__ v16h frag_a(const _Float16* base, int ld, int lane) {
  const _Float16* p = base + (lane & 15) * ld + ((lane >> 4) << 3);
  h8 lo = *reinterpret_cast<const h8*>(p);
  h8 hi = *reinterpret_cast<const h8*>(p + 16);
  return __builtin_shufflevector(lo, hi, 0, 1, 2, 3, 4, 5, 6, 7,
                                 8, 9, 10, 11, 12, 13, 14, 15);
}

// B fragment 32x16 f16, memory holds rows indexed by n with contiguous k:
// lane n holds two 8-half runs [kb, kb+8) and [kb+16, kb+24), kb = (lane>=16)*8 -> two ds_load_b128.
static __device__ __forceinline__ v16h frag_b_nmajor(const _Float16* base, int ld, int lane) {
  const _Float16* p = base + (lane & 15) * ld + ((lane >> 4) << 3);
  h8 lo = *reinterpret_cast<const h8*>(p);
  h8 hi = *reinterpret_cast<const h8*>(p + 16);
  return __builtin_shufflevector(lo, hi, 0, 1, 2, 3, 4, 5, 6, 7,
                                 8, 9, 10, 11, 12, 13, 14, 15);
}

// -------------------------- generic fused GEMM --------------------------
// C[M,N] = act(A[M,K] @ W[N,K]^T + bias[N]) (+ res[(m & res_mask), N])
// M,N multiples of 64; K multiple of 32; res_mask+1 a power of two.
template <bool GELU_EPI>
__global__ __launch_bounds__(128) void gemm_kernel(
    const float* __restrict__ A, const float* __restrict__ Wt,
    const float* __restrict__ bias, const float* __restrict__ res,
    float* __restrict__ C, int M, int N, int K, int res_mask) {
  __shared__ __align__(16) _Float16 As[64 * kLd];
  __shared__ __align__(16) _Float16 Bs[64 * kLd];
  const int tid = threadIdx.x, lane = tid & 31, wave = tid >> 5;
  const int m0 = blockIdx.y * 64, n0 = blockIdx.x * 64;
  const int wm = (wave >> 1) * 32, wn = (wave & 1) * 32;
  const v8f vz = {0.f, 0.f, 0.f, 0.f, 0.f, 0.f, 0.f, 0.f};
  v8f acc[2][2] = {{vz, vz}, {vz, vz}};

  for (int k0 = 0; k0 < K; k0 += 32) {
    __syncthreads();
#pragma unroll
    for (int it = 0; it < 4; ++it) {
      int slot = tid + 128 * it;                // 512 float4 slots per matrix
      int r = slot >> 3, c4 = (slot & 7) << 2;
      float4 va = *reinterpret_cast<const float4*>(&A[(size_t)(m0 + r) * K + k0 + c4]);
      h4 ta; ta[0] = (_Float16)va.x; ta[1] = (_Float16)va.y;
      ta[2] = (_Float16)va.z; ta[3] = (_Float16)va.w;
      *reinterpret_cast<h4*>(&As[r * kLd + c4]) = ta;
      float4 vb = *reinterpret_cast<const float4*>(&Wt[(size_t)(n0 + r) * K + k0 + c4]);
      h4 tb; tb[0] = (_Float16)vb.x; tb[1] = (_Float16)vb.y;
      tb[2] = (_Float16)vb.z; tb[3] = (_Float16)vb.w;
      *reinterpret_cast<h4*>(&Bs[r * kLd + c4]) = tb;
    }
    if (k0 + 32 < K) {                          // CDNA5 global_prefetch of next tile
      int r = tid & 63;
      __builtin_prefetch(&A[(size_t)(m0 + r) * K + k0 + 32], 0, 1);
      __builtin_prefetch(&Wt[(size_t)(n0 + r) * K + k0 + 32], 0, 1);
    }
    __syncthreads();
    v16h a0 = frag_a(As + (wm + 0) * kLd, kLd, lane);
    v16h a1 = frag_a(As + (wm + 16) * kLd, kLd, lane);
    v16h b0 = frag_b_nmajor(Bs + (wn + 0) * kLd, kLd, lane);
    v16h b1 = frag_b_nmajor(Bs + (wn + 16) * kLd, kLd, lane);
    acc[0][0] = wmma_f16(a0, b0, acc[0][0]);
    acc[0][1] = wmma_f16(a0, b1, acc[0][1]);
    acc[1][0] = wmma_f16(a1, b0, acc[1][0]);
    acc[1][1] = wmma_f16(a1, b1, acc[1][1]);
  }
  const int nl = lane & 15, vb8 = (lane >> 4) << 3;
#pragma unroll
  for (int i = 0; i < 2; ++i)
#pragma unroll
    for (int j = 0; j < 2; ++j)
#pragma unroll
      for (int v = 0; v < 8; ++v) {
        int m = m0 + wm + i * 16 + vb8 + v;
        int n = n0 + wn + j * 16 + nl;
        float val = acc[i][j][v] + bias[n];
        if (GELU_EPI) val = 0.5f * val * (1.0f + erff(val * 0.70710678118f));
        if (res) val += res[(size_t)(m & res_mask) * N + n];
        C[(size_t)m * N + n] = val;
      }
}

// -------------------------- layernorm (wave per token) --------------------------
__global__ __launch_bounds__(256) void layernorm_kernel(
    const float* __restrict__ x, const float* __restrict__ w,
    const float* __restrict__ b, float* __restrict__ out, int M) {
  int wave = threadIdx.x >> 5, lane = threadIdx.x & 31;
  int m = blockIdx.x * 8 + wave;
  if (m >= M) return;
  const float* row = x + (size_t)m * kDim;
  float vals[12];
  float s = 0.f;
#pragma unroll
  for (int j = 0; j < 12; ++j) { vals[j] = row[lane + 32 * j]; s += vals[j]; }
#pragma unroll
  for (int o = 16; o > 0; o >>= 1) s += __shfl_xor(s, o, 32);
  float mean = s * (1.0f / kDim);
  float vs = 0.f;
#pragma unroll
  for (int j = 0; j < 12; ++j) { float d = vals[j] - mean; vs += d * d; }
#pragma unroll
  for (int o = 16; o > 0; o >>= 1) vs += __shfl_xor(vs, o, 32);
  float inv = rsqrtf(vs * (1.0f / kDim) + 1e-5f);
  float* orow = out + (size_t)m * kDim;
#pragma unroll
  for (int j = 0; j < 12; ++j) {
    int d = lane + 32 * j;
    orow[d] = (vals[j] - mean) * inv * w[d] + b[d];
  }
}

// -------------------------- bicubic pos-embed resize 14x14 -> 32x32 --------------------------
static __device__ __forceinline__ float cubic_w(float t) {
  t = fabsf(t);
  if (t <= 1.f) return ((1.5f * t - 2.5f) * t) * t + 1.f;
  if (t < 2.f)  return ((-0.5f * t + 2.5f) * t - 4.f) * t + 2.f;
  return 0.f;
}

__global__ __launch_bounds__(128) void pos_resize_kernel(
    const float* __restrict__ pos, float* __restrict__ out) {
  int oy = blockIdx.x >> 5, ox = blockIdx.x & 31;
  float wy[14], wx[14], sy = 0.f, sx = 0.f;
  float cy = (oy + 0.5f) * (14.0f / 32.0f) - 0.5f;
  float cx = (ox + 0.5f) * (14.0f / 32.0f) - 0.5f;
#pragma unroll
  for (int i = 0; i < 14; ++i) {
    wy[i] = cubic_w(cy - i); sy += wy[i];
    wx[i] = cubic_w(cx - i); sx += wx[i];
  }
  float inv = 1.0f / (sy * sx);
  for (int c = threadIdx.x; c < kDim; c += 128) {
    float acc = 0.f;
#pragma unroll
    for (int iy = 0; iy < 14; ++iy) {
      float racc = 0.f;
#pragma unroll
      for (int ix = 0; ix < 14; ++ix)
        racc += wx[ix] * pos[(size_t)(iy * 14 + ix) * kDim + c];
      acc += wy[iy] * racc;
    }
    out[(size_t)blockIdx.x * kDim + c] = acc * inv;
  }
}

// -------------------------- im2col for 16x16/s16 patch conv --------------------------
__global__ __launch_bounds__(128) void im2col_kernel(
    const float* __restrict__ x, float* __restrict__ A) {
  int token = blockIdx.x;
  int b = token >> 10, rem = token & 1023, gy = rem >> 5, gx = rem & 31;
#pragma unroll
  for (int j = 0; j < 6; ++j) {
    int col = threadIdx.x + 128 * j;           // 768 cols: ci*256 + py*16 + px
    int ci = col >> 8, py = (col >> 4) & 15, px = col & 15;
    A[(size_t)token * 768 + col] =
        x[(((size_t)b * 3 + ci) * 512 + gy * 16 + py) * 512 + gx * 16 + px];
  }
}

// -------------------------- window gather / scatter --------------------------
__global__ __launch_bounds__(128) void gather_win_kernel(
    const float* __restrict__ qkv, const float* __restrict__ qkv_b,
    float* __restrict__ qkvw) {
  int w = blockIdx.x / kWN, t = blockIdx.x % kWN;
  int b = w / 9, wrem = w % 9, wy = wrem / 3, wx = wrem % 3;
  int ty = t / kWSz, tx = t % kWSz;
  int hp = wy * kWSz + ty, wp = wx * kWSz + tx;
  bool valid = (hp < kGrid) && (wp < kGrid);
  size_t dst = (size_t)blockIdx.x * kQKVN;
  size_t src = ((size_t)b * kTok + hp * kGrid + wp) * kQKVN;
#pragma unroll
  for (int j = 0; j < 9; ++j) {
    int c = threadIdx.x + 128 * j;
    qkvw[dst + c] = valid ? qkv[src + c] : qkv_b[c];
  }
}

__global__ __launch_bounds__(128) void scatter_win_kernel(
    const float* __restrict__ aw, float* __restrict__ ao) {
  int token = blockIdx.x;
  int b = token >> 10, rem = token & 1023, hq = rem >> 5, wq = rem & 31;
  int w = b * 9 + (hq / kWSz) * 3 + (wq / kWSz);
  int t = (hq % kWSz) * kWSz + (wq % kWSz);
  size_t src = ((size_t)w * kWN + t) * kDim;
  size_t dst = (size_t)token * kDim;
#pragma unroll
  for (int j = 0; j < 3; ++j) {
    int c = threadIdx.x + 128 * j;
    ao[dst + c] = aw[src + c];
  }
}

// -------------------------- decomposed rel-pos bias tables --------------------------
// bias_h[s,h,q,hk] = sum_d q[d]*rel_h[hq-hk+H-1, d];  bias_w analogous.
__global__ __launch_bounds__(64) void rel_bias_kernel(
    const float* __restrict__ qkv, const float* __restrict__ relh,
    const float* __restrict__ relw, float* __restrict__ biasH,
    float* __restrict__ biasW, int N, int H, int W) {
  int qrow = blockIdx.x, head = blockIdx.y, s = blockIdx.z;
  __shared__ float qs[kHD];
  const float* q = qkv + ((size_t)s * N + qrow) * kQKVN + head * kHD;
  if (threadIdx.x < kHD) qs[threadIdx.x] = q[threadIdx.x];
  __syncthreads();
  int t = threadIdx.x;
  size_t rowbase = (size_t)(s * kHeads + head) * N + qrow;
  if (t < 32) {
    if (t < H) {
      int hq = qrow / W;
      const float* r = relh + (size_t)(hq - t + H - 1) * kHD;
      float acc = 0.f;
#pragma unroll
      for (int d = 0; d < kHD; ++d) acc += qs[d] * r[d];
      biasH[rowbase * H + t] = acc;
    }
  } else {
    int wk = t - 32;
    if (wk < W) {
      int wq = qrow % W;
      const float* r = relw + (size_t)(wq - wk + W - 1) * kHD;
      float acc = 0.f;
#pragma unroll
      for (int d = 0; d < kHD; ++d) acc += qs[d] * r[d];
      biasW[rowbase * W + wk] = acc;
    }
  }
}

// -------------------------- flash attention (WMMA, online softmax) --------------------------
// Block = 4 waves, 64 query rows per block; keys streamed in 32-key chunks.
// K staged key-major; V staged TRANSPOSED (d-major) so the P.V B-fragment is
// the contiguous n-major form. FULL=true: N % 64 == 0 (global path, N=1024) ->
// no row/key masking anywhere in the hot loop. FULL=false: branchless
// clamp-and-select masking (no exec divergence) for the windowed N=196 path.
template <bool FULL>
__global__ __launch_bounds__(128) void attn_kernel(
    const float* __restrict__ qkv, const float* __restrict__ biasH,
    const float* __restrict__ biasW, float* __restrict__ out,
    int N, int H, int W) {
  const int head = blockIdx.y, s = blockIdx.z;
  const int tid = threadIdx.x, lane = tid & 31, wave = tid >> 5;
  const int qt0 = blockIdx.x * 64 + wave * 16;
  __shared__ __align__(16) _Float16 Kl[32 * kLd];      // [key][d]
  __shared__ __align__(16) _Float16 Vt[64 * kLd];      // [d][key]  (transposed)
  __shared__ __align__(16) _Float16 Pl[4][16 * kLd];   // per-wave P tile

  const int nl = lane & 15, vb8 = (lane >> 4) << 3;
  const size_t seqbase = (size_t)s * N;
  const size_t brow = (size_t)(s * kHeads + head) * N;
  const v8f vz = {0.f, 0.f, 0.f, 0.f, 0.f, 0.f, 0.f, 0.f};

  // q fragments (pre-scaled by 1/sqrt(d)); two K-steps of 32 over head dim 64.
  // Masked path: clamp the row and zero via select (no exec divergence).
  v16h aq0, aq1;
  {
    int qrow = qt0 + nl;
    bool ok = FULL || (qrow < N);
    int qclamp = FULL ? qrow : (qrow < N ? qrow : N - 1);
    const float* p = qkv + (seqbase + qclamp) * kQKVN + head * kHD;
    int kb = (lane >> 4) << 3;
    float msk = ok ? kScale : 0.f;
#pragma unroll
    for (int half = 0; half < 2; ++half) {
      v16h& f = half ? aq1 : aq0;
      const float* ph = p + half * 32 + kb;
      float4 a0 = *reinterpret_cast<const float4*>(ph);
      float4 a1 = *reinterpret_cast<const float4*>(ph + 4);
      float4 b0 = *reinterpret_cast<const float4*>(ph + 16);
      float4 b1 = *reinterpret_cast<const float4*>(ph + 20);
      f[0]  = (_Float16)(a0.x * msk); f[1]  = (_Float16)(a0.y * msk);
      f[2]  = (_Float16)(a0.z * msk); f[3]  = (_Float16)(a0.w * msk);
      f[4]  = (_Float16)(a1.x * msk); f[5]  = (_Float16)(a1.y * msk);
      f[6]  = (_Float16)(a1.z * msk); f[7]  = (_Float16)(a1.w * msk);
      f[8]  = (_Float16)(b0.x * msk); f[9]  = (_Float16)(b0.y * msk);
      f[10] = (_Float16)(b0.z * msk); f[11] = (_Float16)(b0.w * msk);
      f[12] = (_Float16)(b1.x * msk); f[13] = (_Float16)(b1.y * msk);
      f[14] = (_Float16)(b1.z * msk); f[15] = (_Float16)(b1.w * msk);
    }
  }

  v8f o0 = vz, o1 = vz, o2 = vz, o3 = vz;
  float rm[8], rl[8];
#pragma unroll
  for (int v = 0; v < 8; ++v) { rm[v] = -1e30f; rl[v] = 0.f; }

  for (int kc0 = 0; kc0 < N; kc0 += 32) {
    __syncthreads();
#pragma unroll
    for (int it = 0; it < 4; ++it) {             // stage K,V chunk (32 keys x 64 d)
      int slot = tid + 128 * it;                 // 512 float4 slots
      int r = slot >> 4, d4 = (slot & 15) << 2;
      int kk = kc0 + r;
      bool ok = FULL || (kk < N);
      int kclamp = FULL ? kk : (kk < N ? kk : N - 1);
      const float* pk = qkv + (seqbase + kclamp) * kQKVN + head * kHD + d4;
      float4 kv = *reinterpret_cast<const float4*>(pk + kDim);
      float4 vv = *reinterpret_cast<const float4*>(pk + 2 * kDim);
      float msk = ok ? 1.f : 0.f;
      h4 kt; kt[0] = (_Float16)(kv.x * msk); kt[1] = (_Float16)(kv.y * msk);
      kt[2] = (_Float16)(kv.z * msk); kt[3] = (_Float16)(kv.w * msk);
      *reinterpret_cast<h4*>(&Kl[r * kLd + d4]) = kt;
      Vt[(d4 + 0) * kLd + r] = (_Float16)(vv.x * msk);
      Vt[(d4 + 1) * kLd + r] = (_Float16)(vv.y * msk);
      Vt[(d4 + 2) * kLd + r] = (_Float16)(vv.z * msk);
      Vt[(d4 + 3) * kLd + r] = (_Float16)(vv.w * msk);
    }
    __syncthreads();

    v8f sc[2];
#pragma unroll
    for (int h2 = 0; h2 < 2; ++h2) {             // scores: q(16x64) . k^T(64x16)
      v16h bk0 = frag_b_nmajor(Kl + (h2 * 16) * kLd, kLd, lane);
      v16h bk1 = frag_b_nmajor(Kl + (h2 * 16) * kLd + 32, kLd, lane);
      v8f z = wmma_f16(aq0, bk0, vz);
      sc[h2] = wmma_f16(aq1, bk1, z);
    }
#pragma unroll
    for (int h2 = 0; h2 < 2; ++h2) {             // rel-pos bias + key mask
      int kidx = kc0 + h2 * 16 + nl;
      int hk = kidx / W, wk = kidx % W;
#pragma unroll
      for (int v = 0; v < 8; ++v) {
        int qrow = qt0 + vb8 + v;
        float val = sc[h2][v];
        if (FULL) {
          val += biasH[(brow + qrow) * H + hk] + biasW[(brow + qrow) * W + wk];
        } else {
          if (kidx < N) {
            if (qrow < N)
              val += biasH[(brow + qrow) * H + hk] + biasW[(brow + qrow) * W + wk];
          } else {
            val = -1e30f;
          }
        }
        sc[h2][v] = val;
      }
    }
    float fv[8];
#pragma unroll
    for (int v = 0; v < 8; ++v) {                // online softmax per query row
      float t = fmaxf(sc[0][v], sc[1][v]);
#pragma unroll
      for (int o = 1; o < 16; o <<= 1) t = fmaxf(t, __shfl_xor(t, o, 32));
      float mn = fmaxf(rm[v], t);
      fv[v] = __expf(rm[v] - mn);
      rm[v] = mn;
      float p0 = __expf(sc[0][v] - mn);
      float p1 = __expf(sc[1][v] - mn);
      sc[0][v] = p0; sc[1][v] = p1;
      float ps = p0 + p1;
#pragma unroll
      for (int o = 1; o < 16; o <<= 1) ps += __shfl_xor(ps, o, 32);
      rl[v] = rl[v] * fv[v] + ps;
    }
    _Float16* pw = &Pl[wave][0];                 // re-fragment P via per-wave LDS
#pragma unroll
    for (int h2 = 0; h2 < 2; ++h2)
#pragma unroll
      for (int v = 0; v < 8; ++v)
        pw[(vb8 + v) * kLd + h2 * 16 + nl] = (_Float16)sc[h2][v];
#pragma unroll
    for (int v = 0; v < 8; ++v) {
      o0[v] *= fv[v]; o1[v] *= fv[v]; o2[v] *= fv[v]; o3[v] *= fv[v];
    }
    v16h ap = frag_a(pw, kLd, lane);
    o0 = wmma_f16(ap, frag_b_nmajor(Vt + 0 * 16 * kLd, kLd, lane), o0);
    o1 = wmma_f16(ap, frag_b_nmajor(Vt + 1 * 16 * kLd, kLd, lane), o1);
    o2 = wmma_f16(ap, frag_b_nmajor(Vt + 2 * 16 * kLd, kLd, lane), o2);
    o3 = wmma_f16(ap, frag_b_nmajor(Vt + 3 * 16 * kLd, kLd, lane), o3);
  }
#pragma unroll
  for (int v = 0; v < 8; ++v) {
    int qrow = qt0 + vb8 + v;
    if (FULL || qrow < N) {
      float inv = 1.0f / rl[v];
      float* po = out + (seqbase + qrow) * kDim + head * kHD;
      po[0 + nl]  = o0[v] * inv;
      po[16 + nl] = o1[v] * inv;
      po[32 + nl] = o2[v] * inv;
      po[48 + nl] = o3[v] * inv;
    }
  }
}

// -------------------------- final NHWC -> NCHW --------------------------
__global__ __launch_bounds__(128) void out_transpose_kernel(
    const float* __restrict__ y, float* __restrict__ out) {
  int token = blockIdx.x;
  int b = token >> 10, pix = token & 1023;
#pragma unroll
  for (int j = 0; j < 3; ++j) {
    int c = threadIdx.x + 128 * j;
    out[((size_t)b * kDim + c) * kTok + pix] = y[(size_t)token * kDim + c];
  }
}

// -------------------------- host orchestration --------------------------
extern "C" void kernel_launch(void* const* d_in, const int* in_sizes, int n_in,
                              void* d_out, int out_size, void* d_ws, size_t ws_size,
                              hipStream_t stream) {
  (void)in_sizes; (void)n_in; (void)out_size; (void)ws_size;
  const float* x       = (const float*)d_in[0];
  const float* patch_w = (const float*)d_in[1];   // (384, 768) flattened OIHW
  const float* patch_b = (const float*)d_in[2];
  const float* pos_emb = (const float*)d_in[3];
  const float* n1w = (const float*)d_in[4];
  const float* n1b = (const float*)d_in[5];
  const float* qkv_w = (const float*)d_in[6];
  const float* qkv_b = (const float*)d_in[7];
  const float* proj_w = (const float*)d_in[8];
  const float* proj_b = (const float*)d_in[9];
  const float* n2w = (const float*)d_in[10];
  const float* n2b = (const float*)d_in[11];
  const float* fc1_w = (const float*)d_in[12];
  const float* fc1_b = (const float*)d_in[13];
  const float* fc2_w = (const float*)d_in[14];
  const float* fc2_b = (const float*)d_in[15];
  const float* relh = (const float*)d_in[16];
  const float* relw = (const float*)d_in[17];

  float* ws = (float*)d_ws;
  float* posb = ws + OFF_POS;
  float* im2  = ws + OFF_IM2;
  float* y    = ws + OFF_Y;
  float* nrm  = ws + OFF_NRM;
  float* qkvb = ws + OFF_QKV;
  float* qkvw = ws + OFF_QKVW;
  float* bH   = ws + OFF_BH;
  float* bW   = ws + OFF_BW;
  float* ao   = ws + OFF_AO;
  float* aow  = ws + OFF_AOW;
  float* hbuf = ws + OFF_H;

  // patch embed: im2col GEMM + bias + bicubic pos embed (broadcast over batch)
  pos_resize_kernel<<<kGrid * kGrid, 128, 0, stream>>>(pos_emb, posb);
  im2col_kernel<<<kM, 128, 0, stream>>>(x, im2);
  gemm_kernel<false><<<dim3(kDim / 64, kM / 64), 128, 0, stream>>>(
      im2, patch_w, patch_b, posb, y, kM, kDim, 768, kTok - 1);

  for (int i = 0; i < kDepth; ++i) {
    bool windowed = (i == 0 || i == 1 || i == 3 || i == 4);
    const float* relh_i = relh + (size_t)i * 63 * kHD;
    const float* relw_i = relw + (size_t)i * 63 * kHD;

    layernorm_kernel<<<kM / 8, 256, 0, stream>>>(y, n1w + i * kDim, n1b + i * kDim, nrm, kM);
    gemm_kernel<false><<<dim3(kQKVN / 64, kM / 64), 128, 0, stream>>>(
        nrm, qkv_w + (size_t)i * kQKVN * kDim, qkv_b + i * kQKVN, nullptr,
        qkvb, kM, kQKVN, kDim, kM - 1);

    if (windowed) {
      gather_win_kernel<<<kNWin * kWN, 128, 0, stream>>>(qkvb, qkv_b + i * kQKVN, qkvw);
      rel_bias_kernel<<<dim3(kWN, kHeads, kNWin), 64, 0, stream>>>(
          qkvw, relh_i, relw_i, bH, bW, kWN, kWSz, kWSz);
      attn_kernel<false><<<dim3((kWN + 63) / 64, kHeads, kNWin), 128, 0, stream>>>(
          qkvw, bH, bW, aow, kWN, kWSz, kWSz);
      scatter_win_kernel<<<kM, 128, 0, stream>>>(aow, ao);
    } else {
      rel_bias_kernel<<<dim3(kTok, kHeads, kBatch), 64, 0, stream>>>(
          qkvb, relh_i, relw_i, bH, bW, kTok, kGrid, kGrid);
      attn_kernel<true><<<dim3(kTok / 64, kHeads, kBatch), 128, 0, stream>>>(
          qkvb, bH, bW, ao, kTok, kGrid, kGrid);
    }

    // y = y + proj(attn)   (residual read/write same row -> in-place safe)
    gemm_kernel<false><<<dim3(kDim / 64, kM / 64), 128, 0, stream>>>(
        ao, proj_w + (size_t)i * kDim * kDim, proj_b + i * kDim, y, y,
        kM, kDim, kDim, kM - 1);

    layernorm_kernel<<<kM / 8, 256, 0, stream>>>(y, n2w + i * kDim, n2b + i * kDim, nrm, kM);
    gemm_kernel<true><<<dim3(1536 / 64, kM / 64), 128, 0, stream>>>(
        nrm, fc1_w + (size_t)i * 1536 * kDim, fc1_b + i * 1536, nullptr,
        hbuf, kM, 1536, kDim, kM - 1);
    gemm_kernel<false><<<dim3(kDim / 64, kM / 64), 128, 0, stream>>>(
        hbuf, fc2_w + (size_t)i * kDim * 1536, fc2_b + i * kDim, y, y,
        kM, kDim, 1536, kM - 1);
  }

  out_transpose_kernel<<<kM, 128, 0, stream>>>(y, (float*)d_out);
}
